// RwkvForClassification_24343874634276
// MI455X (gfx1250) — compile-verified, matrix operand order checked by reading, not run on hardware
//
#include <hip/hip_runtime.h>
#include <hip/hip_bf16.h>

// Problem constants (from the reference)
#define T_  2048
#define D_  1024
#define L_  12
#define H_  16
#define DH_ 64
#define F_  3584
#define NL_ 2

typedef __attribute__((ext_vector_type(16))) __bf16 v16bf;
typedef __attribute__((ext_vector_type(8)))  float  v8f;

// Pack two f32 -> packed bf16 pair (RNE), single v_cvt_pk_bf16_f32.
// (Mnemonic confirmed in gfx1250 codegen from the previous round.)
__device__ __forceinline__ unsigned int pk_bf16(float lo, float hi) {
  unsigned int u;
  asm("v_cvt_pk_bf16_f32 %0, %1, %2" : "=v"(u) : "v"(lo), "v"(hi));
  return u;
}

// ---------------------------------------------------------------------------
// bf16 WMMA GEMM: C[M,N] (f32) = act(A[M,K] @ B[K,N]) (+C if addC)
// f32 operands converted to bf16 ONCE while staging into LDS; fragment loads
// are pure ds_load_b128 per the 16-bit WMMA A/B VGPR layout.
// Tiles: 128x128 C per 256-thread block, BK=64 (2 WMMA k-steps per stage).
// 8 waves each own 32x64 of C (2x4 16x16 f32 accumulators).
// Staging: batch ALL global loads to registers BEFORE the barrier so their
// latency overlaps the barrier wait; convert + ds_store afterwards.
// ---------------------------------------------------------------------------
#define BM 128
#define BN 128
#define BK 64
#define LDSP 72   // padded LDS row stride in bf16 elems (144B: 16B-aligned)

__global__ __launch_bounds__(256)
void k_gemm_bf16(const float* __restrict__ A, const float* __restrict__ B,
                 float* __restrict__ C, int M, int N, int K, int addC, int act)
{
  __shared__ __align__(16) unsigned short sA[BM * LDSP];  // sA[row][k] bf16
  __shared__ __align__(16) unsigned short sB[BN * LDSP];  // sB[n][k]  bf16 (transposed)

  const int tid  = threadIdx.x;
  const int half = (tid >> 4) & 1;    // lane>>4
  const int r16  = tid & 15;          // lane&15
  const int wid  = tid >> 5;
  const int waveM = (wid >> 1) * 32;  // 4 wave-rows
  const int waveN = (wid & 1) * 64;   // 2 wave-cols
  const int bM = blockIdx.y * BM;
  const int bN = blockIdx.x * BN;

  // A-staging coords
  const int rowA0 = tid >> 4;         // + p*16
  const int colA  = (tid & 15) * 4;
  // B-staging coords: each thread owns column n, half of the k-range
  const int nB  = tid & 127;
  const int kqB = (tid >> 7) * 32;

  v8f acc[2][4];
  #pragma unroll
  for (int i2 = 0; i2 < 2; ++i2)
    #pragma unroll
    for (int j2 = 0; j2 < 4; ++j2)
      #pragma unroll
      for (int e = 0; e < 8; ++e) acc[i2][j2][e] = 0.f;

  union BFu { v16bf v; uint4 q[2]; };

  for (int k0 = 0; k0 < K; k0 += BK) {
    // ---- batch-issue ALL tile loads into registers (latency overlaps barrier)
    float4 aReg[8];
    #pragma unroll
    for (int p = 0; p < 8; ++p) {
      const int row = rowA0 + p * 16;
      aReg[p] = *reinterpret_cast<const float4*>(
          &A[(size_t)(bM + row) * K + k0 + colA]);
    }
    float bReg[32];
    #pragma unroll
    for (int p = 0; p < 8; ++p) {
      const int kk = kqB + p * 4;
      const float* bp = &B[(size_t)(k0 + kk) * N + bN + nB];
      bReg[4 * p + 0] = bp[0];
      bReg[4 * p + 1] = bp[(size_t)N];
      bReg[4 * p + 2] = bp[(size_t)2 * N];
      bReg[4 * p + 3] = bp[(size_t)3 * N];
    }
    // prefetch next K-tile into the cache hierarchy (global_prefetch_b8)
    if (k0 + BK < K) {
      __builtin_prefetch(&A[(size_t)(bM + rowA0) * K + k0 + BK], 0, 1);
      __builtin_prefetch(&B[(size_t)(k0 + BK + kqB) * N + bN + nB], 0, 1);
    }

    __syncthreads();   // previous iteration's fragment reads complete

    // ---- convert + store to LDS (bf16, packed pairs) ----
    #pragma unroll
    for (int p = 0; p < 8; ++p) {
      const int row = rowA0 + p * 16;
      uint2 w;
      w.x = pk_bf16(aReg[p].x, aReg[p].y);
      w.y = pk_bf16(aReg[p].z, aReg[p].w);
      *reinterpret_cast<uint2*>(&sA[row * LDSP + colA]) = w;
    }
    #pragma unroll
    for (int p = 0; p < 8; ++p) {
      const int kk = kqB + p * 4;
      uint2 w;
      w.x = pk_bf16(bReg[4 * p + 0], bReg[4 * p + 1]);
      w.y = pk_bf16(bReg[4 * p + 2], bReg[4 * p + 3]);
      *reinterpret_cast<uint2*>(&sB[nB * LDSP + kk]) = w;
    }
    __syncthreads();

    // ---- 2 WMMA k-steps; fragments are pure 16B ds loads (no VALU convert).
    //      16-bit A layout: lane half h holds k = h*8+{0..7} (elems 0-7) and
    //      16+h*8+{0..7} (elems 8-15) within each 32-wide k-step.
    #pragma unroll
    for (int ks = 0; ks < 2; ++ks) {
      const int kb = ks * 32 + half * 8;
      BFu afr[2], bfr[4];
      #pragma unroll
      for (int i2 = 0; i2 < 2; ++i2) {
        const unsigned short* s = &sA[(waveM + i2 * 16 + r16) * LDSP + kb];
        afr[i2].q[0] = *reinterpret_cast<const uint4*>(s);
        afr[i2].q[1] = *reinterpret_cast<const uint4*>(s + 16);
      }
      #pragma unroll
      for (int j2 = 0; j2 < 4; ++j2) {
        const unsigned short* s = &sB[(waveN + j2 * 16 + r16) * LDSP + kb];
        bfr[j2].q[0] = *reinterpret_cast<const uint4*>(s);
        bfr[j2].q[1] = *reinterpret_cast<const uint4*>(s + 16);
      }
      #pragma unroll
      for (int i2 = 0; i2 < 2; ++i2)
        #pragma unroll
        for (int j2 = 0; j2 < 4; ++j2)
          acc[i2][j2] = __builtin_amdgcn_wmma_f32_16x16x32_bf16(
              false, afr[i2].v, false, bfr[j2].v, (short)0, acc[i2][j2],
              false, false);
    }
  }

  // ---- epilogue: C/D layout -> m = j + half*8, n = lane&15 ----
  #pragma unroll
  for (int i2 = 0; i2 < 2; ++i2)
    #pragma unroll
    for (int j2 = 0; j2 < 4; ++j2)
      #pragma unroll
      for (int j = 0; j < 8; ++j) {
        const int m = bM + waveM + i2 * 16 + half * 8 + j;
        const int n = bN + waveN + j2 * 16 + r16;
        float v = acc[i2][j2][j];
        if (act == 1) { v = (v > 0.f) ? v * v : 0.f; }          // relu^2
        else if (act == 2) { v = 1.f / (1.f + __expf(-v)); }     // sigmoid
        const size_t off = (size_t)m * N + n;
        if (addC) v += C[off];
        C[off] = v;
      }
}

// ---------------------------------------------------------------------------
// Embedding gather + LayerNorm (ln0). One block per token, 256 thr, D=1024.
// ---------------------------------------------------------------------------
__global__ __launch_bounds__(256)
void k_embed_ln(const int* __restrict__ idx, const float* __restrict__ emb,
                const float* __restrict__ w, const float* __restrict__ b,
                float* __restrict__ out)
{
  __shared__ float red[256];
  __shared__ float s_m, s_r;
  const int tid = threadIdx.x, t = blockIdx.x;
  const float* e = emb + (size_t)idx[t] * D_;
  float v[4]; float s = 0.f;
  #pragma unroll
  for (int i = 0; i < 4; ++i) { v[i] = e[i * 256 + tid]; s += v[i]; }
  red[tid] = s; __syncthreads();
  for (int st = 128; st > 0; st >>= 1) { if (tid < st) red[tid] += red[tid + st]; __syncthreads(); }
  if (tid == 0) s_m = red[0] * (1.f / D_);
  __syncthreads();
  float ss = 0.f;
  #pragma unroll
  for (int i = 0; i < 4; ++i) { float d = v[i] - s_m; ss += d * d; }
  red[tid] = ss; __syncthreads();
  for (int st = 128; st > 0; st >>= 1) { if (tid < st) red[tid] += red[tid + st]; __syncthreads(); }
  if (tid == 0) s_r = rsqrtf(red[0] * (1.f / D_) + 1e-5f);
  __syncthreads();
  #pragma unroll
  for (int i = 0; i < 4; ++i) {
    const int j = i * 256 + tid;
    out[(size_t)t * D_ + j] = (v[i] - s_m) * s_r * w[j] + b[j];
  }
}

// ---------------------------------------------------------------------------
// LayerNorm over D=1024. One block per token.
// ---------------------------------------------------------------------------
__global__ __launch_bounds__(256)
void k_layernorm(const float* __restrict__ in, const float* __restrict__ w,
                 const float* __restrict__ b, float* __restrict__ out)
{
  __shared__ float red[256];
  __shared__ float s_m, s_r;
  const int tid = threadIdx.x, t = blockIdx.x;
  const float* x = in + (size_t)t * D_;
  float v[4]; float s = 0.f;
  #pragma unroll
  for (int i = 0; i < 4; ++i) { v[i] = x[i * 256 + tid]; s += v[i]; }
  red[tid] = s; __syncthreads();
  for (int st = 128; st > 0; st >>= 1) { if (tid < st) red[tid] += red[tid + st]; __syncthreads(); }
  if (tid == 0) s_m = red[0] * (1.f / D_);
  __syncthreads();
  float ss = 0.f;
  #pragma unroll
  for (int i = 0; i < 4; ++i) { float d = v[i] - s_m; ss += d * d; }
  red[tid] = ss; __syncthreads();
  for (int st = 128; st > 0; st >>= 1) { if (tid < st) red[tid] += red[tid + st]; __syncthreads(); }
  if (tid == 0) s_r = rsqrtf(red[0] * (1.f / D_) + 1e-5f);
  __syncthreads();
  #pragma unroll
  for (int i = 0; i < 4; ++i) {
    const int j = i * 256 + tid;
    out[(size_t)t * D_ + j] = (v[i] - s_m) * s_r * w[j] + b[j];
  }
}

// ---------------------------------------------------------------------------
// Token-shift mixes (4-way and 2-way): out = xn*m + xn_prev*(1-m)
// ---------------------------------------------------------------------------
__global__ void k_mix4(const float* __restrict__ xn,
                       const float* __restrict__ mr, const float* __restrict__ mk,
                       const float* __restrict__ mv, const float* __restrict__ mg,
                       float* __restrict__ xr, float* __restrict__ xk,
                       float* __restrict__ xv, float* __restrict__ xg)
{
  const int i = blockIdx.x * blockDim.x + threadIdx.x;
  if (i >= T_ * D_) return;
  const int j = i & (D_ - 1);
  const float cur = xn[i];
  const float prev = (i >= D_) ? xn[i - D_] : 0.f;
  float a;
  a = mr[j]; xr[i] = cur * a + prev * (1.f - a);
  a = mk[j]; xk[i] = cur * a + prev * (1.f - a);
  a = mv[j]; xv[i] = cur * a + prev * (1.f - a);
  a = mg[j]; xg[i] = cur * a + prev * (1.f - a);
}

__global__ void k_mix2(const float* __restrict__ xn,
                       const float* __restrict__ mk, const float* __restrict__ mr,
                       float* __restrict__ xk, float* __restrict__ xr)
{
  const int i = blockIdx.x * blockDim.x + threadIdx.x;
  if (i >= T_ * D_) return;
  const int j = i & (D_ - 1);
  const float cur = xn[i];
  const float prev = (i >= D_) ? xn[i - D_] : 0.f;
  float a;
  a = mk[j]; xk[i] = cur * a + prev * (1.f - a);
  a = mr[j]; xr[i] = cur * a + prev * (1.f - a);
}

// ---------------------------------------------------------------------------
// RWKV-5 WKV recurrence, one block per head (16 blocks x 256 threads).
// State S[64][64] distributed: thread (dg,e) owns S[dg*16+dl][e], dl=0..15.
// Per-thread decay/bonus slices live in registers; r/k are read as float4
// broadcast LDS loads (all lanes of a wave share dg -> conflict-free b128).
// ---------------------------------------------------------------------------
__global__ __launch_bounds__(256)
void k_wkv(const float* __restrict__ r, const float* __restrict__ k,
           const float* __restrict__ v, const float* __restrict__ decay,
           const float* __restrict__ faaaa, float* __restrict__ out)
{
  __shared__ __align__(16) float rs[DH_];
  __shared__ __align__(16) float ks[DH_];
  __shared__ __align__(16) float vs[DH_];
  __shared__ float part[4][DH_];
  const int tid = threadIdx.x;
  const int h = blockIdx.x;
  const int e = tid & (DH_ - 1);
  const int dg = tid >> 6;           // 0..3

  float uR[16], wR[16], S[16];
  #pragma unroll
  for (int dl = 0; dl < 16; ++dl) {
    const int d = dg * 16 + dl;
    uR[dl] = faaaa[h * DH_ + d];
    wR[dl] = __expf(-__expf(decay[h * DH_ + d]));
    S[dl] = 0.f;
  }

  for (int t = 0; t < T_; ++t) {
    const int base = t * D_ + h * DH_;
    if (tid < 64)       rs[tid]        = r[base + tid];
    else if (tid < 128) ks[tid - 64]   = k[base + tid - 64];
    else if (tid < 192) vs[tid - 128]  = v[base + tid - 128];
    __syncthreads();

    const float ve = vs[e];
    float acc = 0.f;
    #pragma unroll
    for (int q = 0; q < 4; ++q) {
      const float4 k4 = *reinterpret_cast<const float4*>(&ks[dg * 16 + q * 4]);
      const float4 r4 = *reinterpret_cast<const float4*>(&rs[dg * 16 + q * 4]);
      const int dl = q * 4;
      float kv;
      kv = k4.x * ve; acc += r4.x * (uR[dl+0]*kv + S[dl+0]); S[dl+0] = wR[dl+0]*S[dl+0] + kv;
      kv = k4.y * ve; acc += r4.y * (uR[dl+1]*kv + S[dl+1]); S[dl+1] = wR[dl+1]*S[dl+1] + kv;
      kv = k4.z * ve; acc += r4.z * (uR[dl+2]*kv + S[dl+2]); S[dl+2] = wR[dl+2]*S[dl+2] + kv;
      kv = k4.w * ve; acc += r4.w * (uR[dl+3]*kv + S[dl+3]); S[dl+3] = wR[dl+3]*S[dl+3] + kv;
    }
    part[dg][e] = acc;
    __syncthreads();
    if (dg == 0)
      out[base + e] = part[0][e] + part[1][e] + part[2][e] + part[3][e];
  }
}

// ---------------------------------------------------------------------------
// Per-head GroupNorm of wkv, affine over D, fused with silu(g) multiply.
// One block per (t, h); 64 threads.
// ---------------------------------------------------------------------------
__global__ __launch_bounds__(64)
void k_gn_silu_mul(const float* __restrict__ wkv, const float* __restrict__ g,
                   const float* __restrict__ lw, const float* __restrict__ lb,
                   float* __restrict__ out)
{
  __shared__ float red[64];
  __shared__ float s_m, s_r;
  const int t = blockIdx.x / H_, h = blockIdx.x % H_;
  const int e = threadIdx.x;
  const float val = wkv[(size_t)t * D_ + h * DH_ + e];
  red[e] = val; __syncthreads();
  for (int st = 32; st > 0; st >>= 1) { if (e < st) red[e] += red[e + st]; __syncthreads(); }
  if (e == 0) s_m = red[0] * (1.f / DH_);
  __syncthreads();
  const float d0 = val - s_m;
  red[e] = d0 * d0; __syncthreads();
  for (int st = 32; st > 0; st >>= 1) { if (e < st) red[e] += red[e + st]; __syncthreads(); }
  if (e == 0) s_r = rsqrtf(red[0] * (1.f / DH_) + 1e-5f);
  __syncthreads();
  const int j = h * DH_ + e;
  const float gn = d0 * s_r * lw[j] + lb[j];
  const float gv = g[(size_t)t * D_ + j];
  out[(size_t)t * D_ + j] = gn * (gv / (1.f + __expf(-gv)));   // * silu(g)
}

// x += a * b
__global__ void k_mul_add(float* __restrict__ x, const float* __restrict__ a,
                          const float* __restrict__ b)
{
  const int i = blockIdx.x * blockDim.x + threadIdx.x;
  if (i < T_ * D_) x[i] += a[i] * b[i];
}

// ---------------------------------------------------------------------------
// Final LN(lnout) on last token + classifier head (NL=2 dot products).
// ---------------------------------------------------------------------------
__global__ __launch_bounds__(256)
void k_head(const float* __restrict__ x, const float* __restrict__ w,
            const float* __restrict__ b, const float* __restrict__ score,
            float* __restrict__ out)
{
  __shared__ float red[256];
  __shared__ float s_m, s_r;
  const int tid = threadIdx.x;
  const float* xr = x + (size_t)(T_ - 1) * D_;
  float v[4]; float s = 0.f;
  #pragma unroll
  for (int i = 0; i < 4; ++i) { v[i] = xr[i * 256 + tid]; s += v[i]; }
  red[tid] = s; __syncthreads();
  for (int st = 128; st > 0; st >>= 1) { if (tid < st) red[tid] += red[tid + st]; __syncthreads(); }
  if (tid == 0) s_m = red[0] * (1.f / D_);
  __syncthreads();
  float ss = 0.f;
  #pragma unroll
  for (int i = 0; i < 4; ++i) { float d = v[i] - s_m; ss += d * d; }
  red[tid] = ss; __syncthreads();
  for (int st = 128; st > 0; st >>= 1) { if (tid < st) red[tid] += red[tid + st]; __syncthreads(); }
  if (tid == 0) s_r = rsqrtf(red[0] * (1.f / D_) + 1e-5f);
  __syncthreads();
  float d0 = 0.f, d1 = 0.f;
  #pragma unroll
  for (int i = 0; i < 4; ++i) {
    const int j = i * 256 + tid;
    const float n = (v[i] - s_m) * s_r * w[j] + b[j];
    d0 += n * score[j];
    d1 += n * score[D_ + j];
  }
  red[tid] = d0; __syncthreads();
  for (int st = 128; st > 0; st >>= 1) { if (tid < st) red[tid] += red[tid + st]; __syncthreads(); }
  if (tid == 0) out[0] = red[0];
  __syncthreads();
  red[tid] = d1; __syncthreads();
  for (int st = 128; st > 0; st >>= 1) { if (tid < st) red[tid] += red[tid + st]; __syncthreads(); }
  if (tid == 0) out[1] = red[0];
}

// ---------------------------------------------------------------------------
// Host orchestration
// ---------------------------------------------------------------------------
extern "C" void kernel_launch(void* const* d_in, const int* in_sizes, int n_in,
                              void* d_out, int out_size, void* d_ws, size_t ws_size,
                              hipStream_t stream)
{
  (void)in_sizes; (void)n_in; (void)out_size; (void)ws_size;

  const int*   idx     = (const int*)  d_in[0];
  const float* emb     = (const float*)d_in[1];
  const float* ln0_w   = (const float*)d_in[2];
  const float* ln0_b   = (const float*)d_in[3];
  const float* ln1_w   = (const float*)d_in[4];
  const float* ln1_b   = (const float*)d_in[5];
  const float* mix_r   = (const float*)d_in[6];
  const float* mix_k   = (const float*)d_in[7];
  const float* mix_v   = (const float*)d_in[8];
  const float* mix_g   = (const float*)d_in[9];
  const float* decay   = (const float*)d_in[10];
  const float* faaaa   = (const float*)d_in[11];
  const float* Wr      = (const float*)d_in[12];
  const float* Wk      = (const float*)d_in[13];
  const float* Wv      = (const float*)d_in[14];
  const float* Wg      = (const float*)d_in[15];
  const float* Wo      = (const float*)d_in[16];
  const float* lnx_w   = (const float*)d_in[17];
  const float* lnx_b   = (const float*)d_in[18];
  const float* ln2_w   = (const float*)d_in[19];
  const float* ln2_b   = (const float*)d_in[20];
  const float* cmix_k  = (const float*)d_in[21];
  const float* cmix_r  = (const float*)d_in[22];
  const float* cWk     = (const float*)d_in[23];
  const float* cWv     = (const float*)d_in[24];
  const float* cWr     = (const float*)d_in[25];
  const float* lnout_w = (const float*)d_in[26];
  const float* lnout_b = (const float*)d_in[27];
  const float* score_w = (const float*)d_in[28];

  float* ws = (float*)d_ws;
  const size_t TD = (size_t)T_ * D_;
  float* x  = ws;            // residual stream
  float* s0 = x  + TD;       // xn / xn2
  float* s1 = s0 + TD;       // xr / wkv / xk2 / kv
  float* s2 = s1 + TD;       // xk / gn*g / rr
  float* s3 = s2 + TD;       // xv / xr2
  float* s4 = s3 + TD;       // xg
  float* s5 = s4 + TD;       // r
  float* s6 = s5 + TD;       // k
  float* s7 = s6 + TD;       // v
  float* s8 = s7 + TD;       // g (pre-silu)
  float* s9 = s8 + TD;       // kk (T x F)

  const int EW = (T_ * D_) / 256;       // elementwise grid
  const dim3 gDD(D_ / BN, T_ / BM);     // GEMM grids
  const dim3 gDF(F_ / BN, T_ / BM);

  k_embed_ln<<<T_, 256, 0, stream>>>(idx, emb, ln0_w, ln0_b, x);

  for (int i = 0; i < L_; ++i) {
    const size_t oD  = (size_t)i * D_;
    const size_t oDD = (size_t)i * D_ * D_;
    const size_t oHD = (size_t)i * H_ * DH_;

    // ---- time mixing ----
    k_layernorm<<<T_, 256, 0, stream>>>(x, ln1_w + oD, ln1_b + oD, s0);
    k_mix4<<<EW, 256, 0, stream>>>(s0, mix_r + oD, mix_k + oD, mix_v + oD,
                                   mix_g + oD, s1, s2, s3, s4);
    k_gemm_bf16<<<gDD, 256, 0, stream>>>(s1, Wr + oDD, s5, T_, D_, D_, 0, 0);
    k_gemm_bf16<<<gDD, 256, 0, stream>>>(s2, Wk + oDD, s6, T_, D_, D_, 0, 0);
    k_gemm_bf16<<<gDD, 256, 0, stream>>>(s3, Wv + oDD, s7, T_, D_, D_, 0, 0);
    k_gemm_bf16<<<gDD, 256, 0, stream>>>(s4, Wg + oDD, s8, T_, D_, D_, 0, 0);
    k_wkv<<<H_, 256, 0, stream>>>(s5, s6, s7, decay + oHD, faaaa + oHD, s1);
    k_gn_silu_mul<<<T_ * H_, 64, 0, stream>>>(s1, s8, lnx_w + oD, lnx_b + oD, s2);
    k_gemm_bf16<<<gDD, 256, 0, stream>>>(s2, Wo + oDD, x, T_, D_, D_, 1, 0);

    // ---- channel mixing ----
    k_layernorm<<<T_, 256, 0, stream>>>(x, ln2_w + oD, ln2_b + oD, s0);
    k_mix2<<<EW, 256, 0, stream>>>(s0, cmix_k + oD, cmix_r + oD, s1, s3);
    k_gemm_bf16<<<gDF, 256, 0, stream>>>(s1, cWk + (size_t)i * D_ * F_, s9,
                                         T_, F_, D_, 0, 1);            // relu^2
    k_gemm_bf16<<<gDD, 256, 0, stream>>>(s3, cWr + oDD, s2,
                                         T_, D_, D_, 0, 2);            // sigmoid
    k_gemm_bf16<<<gDD, 256, 0, stream>>>(s9, cWv + (size_t)i * F_ * D_, s1,
                                         T_, D_, F_, 0, 0);
    k_mul_add<<<EW, 256, 0, stream>>>(x, s2, s1);
  }

  k_head<<<1, 256, 0, stream>>>(x, lnout_w, lnout_b, score_w, (float*)d_out);
}
